// ScaledDotProductCrossAttention_56032143344236
// MI455X (gfx1250) — matrix operand order repaired and checked by reading, not compiled, not run on hardware
//
#include <hip/hip_runtime.h>
#include <hip/hip_bf16.h>

// ---------------------------------------------------------------------------
// CDNA5 / gfx1250 wave32 WMMA implementation of grouped cross-attention.
// All matrix contractions use v_wmma_f32_16x16x32_bf16 (bf16 in, f32 acc).
// V tiles in attention are staged HBM -> LDS with GLOBAL_LOAD_ASYNC_TO_LDS_B128
// (ASYNCcnt), issued before the score WMMAs so the DMA overlaps matrix math.
//
// Fragment layouts follow CDNA5 ISA section 7.12.2:
//   A (16x32 bf16):  lane L holds row M=L%16;
//       u[0..3] : K = kb + (L>=16?8:0)  + {0..7}   (8 contiguous bf16)
//       u[4..7] : K = kb + 16 + (L>=16?8:0) + {0..7}
//   B (32x16 bf16):  lane L holds col N=L%16;
//       u[0..3] : K = kb + (L>=16?8:0) + {0..7}; u[4..7] : K = kb + 16 + (L>=16?8:0) + {0..7}
//   C/D (16x16 f32): lane L holds col N=L%16; VGPR v = row (v + (L>=16?8:0))
// ---------------------------------------------------------------------------

#define BB 2
#define TQL 512
#define TKL 8192
#define DDIM 1024
#define NHEAD 16
#define HDIM 64

typedef __attribute__((ext_vector_type(16))) __bf16 v16bf;
typedef __attribute__((ext_vector_type(8)))  float  v8f;
typedef __attribute__((ext_vector_type(4)))  unsigned int u32x4;

union Frag {
  v16bf v;
  u32x4 q[2];
  unsigned int u[8];
  unsigned short h[16];
};

__device__ __forceinline__ unsigned short f2bf(float f) {
  unsigned int u = __float_as_uint(f);
  u += 0x7FFFu + ((u >> 16) & 1u);   // round-to-nearest-even
  return (unsigned short)(u >> 16);
}

__device__ __forceinline__ float rmax16(float x) {
#pragma unroll
  for (int m = 1; m < 16; m <<= 1) x = fmaxf(x, __shfl_xor(x, m, 32));
  return x;
}
__device__ __forceinline__ float rsum16(float x) {
#pragma unroll
  for (int m = 1; m < 16; m <<= 1) x += __shfl_xor(x, m, 32);
  return x;
}

// ------------------- fp32 -> bf16 conversion (8 elems/thread) --------------
__global__ void k_cvt8(const float* __restrict__ src,
                       unsigned short* __restrict__ dst, long n8) {
  long i = (long)blockIdx.x * blockDim.x + threadIdx.x;   // index of 8-group
  if (i >= n8) return;
  const float4* s4 = (const float4*)(src) + i * 2;
  float4 f0 = s4[0];
  float4 f1 = s4[1];
  u32x4 r;
  r.x = (unsigned)f2bf(f0.x) | ((unsigned)f2bf(f0.y) << 16);
  r.y = (unsigned)f2bf(f0.z) | ((unsigned)f2bf(f0.w) << 16);
  r.z = (unsigned)f2bf(f1.x) | ((unsigned)f2bf(f1.y) << 16);
  r.w = (unsigned)f2bf(f1.z) | ((unsigned)f2bf(f1.w) << 16);
  ((u32x4*)dst)[i] = r;
}

// kv + (row<Na ? active_bias : inactive_bias), then bf16 (8 elems/thread)
__global__ void k_cvt_kv8(const float* __restrict__ kv,
                          const float* __restrict__ ab,
                          const float* __restrict__ ib,
                          const int* __restrict__ nap,
                          unsigned short* __restrict__ dst, long n8) {
  long i = (long)blockIdx.x * blockDim.x + threadIdx.x;
  if (i >= n8) return;
  long e0 = i * 8;
  int col = (int)(e0 % DDIM);
  long row = e0 / DDIM;
  int t = (int)(row % TKL);
  const float* bsrc = (t < *nap) ? ab : ib;
  const float4* b4 = (const float4*)(bsrc + col);
  const float4* s4 = (const float4*)(kv + e0);
  float4 f0 = s4[0], f1 = s4[1];
  float4 g0 = b4[0], g1 = b4[1];
  u32x4 r;
  r.x = (unsigned)f2bf(f0.x + g0.x) | ((unsigned)f2bf(f0.y + g0.y) << 16);
  r.y = (unsigned)f2bf(f0.z + g0.z) | ((unsigned)f2bf(f0.w + g0.w) << 16);
  r.z = (unsigned)f2bf(f1.x + g1.x) | ((unsigned)f2bf(f1.y + g1.y) << 16);
  r.w = (unsigned)f2bf(f1.z + g1.z) | ((unsigned)f2bf(f1.w + g1.w) << 16);
  ((u32x4*)dst)[i] = r;
}

// ------------------- bf16 WMMA GEMM: C = A(MxK) * W(NxK)^T + bias ----------
// One wave computes a 32x64 tile; 8 WMMAs per k-step of 32.
// If log_temp != nullptr: scale result by 1/(64*clip(exp(lt),0.1,10)).
__global__ void k_gemm(const unsigned short* __restrict__ A,
                       const unsigned short* __restrict__ W,
                       const float* __restrict__ bias,
                       const float* __restrict__ log_temp,
                       unsigned short* __restrict__ Cb,
                       float* __restrict__ Cf,
                       int M, int N, int K) {
  const int lane = threadIdx.x;
  const int half = lane >> 4;
  const int lan  = lane & 15;
  const int kh8  = half * 8;
  const int kh16 = half * 16;
  const int n0 = blockIdx.x * 64;
  const int m0 = blockIdx.y * 32;

  const v8f z = {0.f, 0.f, 0.f, 0.f, 0.f, 0.f, 0.f, 0.f};
  v8f acc[2][4];
#pragma unroll
  for (int i = 0; i < 2; ++i)
#pragma unroll
    for (int t = 0; t < 4; ++t) acc[i][t] = z;

  const unsigned short* pa0 = A + (long)(m0 + lan) * K + kh8;
  const unsigned short* pa1 = pa0 + 16L * K;
  const unsigned short* pw0 = W + (long)(n0 + lan) * K + kh8;

  for (int kb = 0; kb < K; kb += 32) {
    Frag a[2], bf[4];
    a[0].q[0] = *(const u32x4*)(pa0 + kb);
    a[0].q[1] = *(const u32x4*)(pa0 + kb + 16);
    a[1].q[0] = *(const u32x4*)(pa1 + kb);
    a[1].q[1] = *(const u32x4*)(pa1 + kb + 16);
#pragma unroll
    for (int t = 0; t < 4; ++t) {
      const unsigned short* p = pw0 + (long)(16 * t) * K + kb;
      bf[t].q[0] = *(const u32x4*)(p);
      bf[t].q[1] = *(const u32x4*)(p + 16);
    }
#pragma unroll
    for (int i = 0; i < 2; ++i)
#pragma unroll
      for (int t = 0; t < 4; ++t)
        acc[i][t] = __builtin_amdgcn_wmma_f32_16x16x32_bf16(
            false, a[i].v, false, bf[t].v, (short)0, acc[i][t], false, false);
  }

  float scale = 1.0f;
  if (log_temp) {
    float tm = __expf(*log_temp);
    tm = fminf(fmaxf(tm, 0.1f), 10.0f);
    scale = 1.0f / (64.0f * tm);   // HEAD_DIM * temp
  }

#pragma unroll
  for (int i = 0; i < 2; ++i)
#pragma unroll
    for (int t = 0; t < 4; ++t) {
      int c = n0 + 16 * t + lan;
      float bv = bias ? bias[c] : 0.0f;
#pragma unroll
      for (int v = 0; v < 8; ++v) {
        int r = m0 + 16 * i + v + kh8;
        float val = (acc[i][t][v] + bv) * scale;
        long idx = (long)r * N + c;
        if (Cb) Cb[idx] = f2bf(val);
        else    Cf[idx] = val;
      }
    }
}

// ------------------ flash-style two-group attention ------------------------
// One wave per (b, h, 16-query tile). Keys in chunks of 32.
// out = attn(active)/sqrt(Na) - attn(inactive)/sqrt(Ni)
__global__ void k_attn(const unsigned short* __restrict__ Q,   // scaled q proj
                       const unsigned short* __restrict__ Kp,
                       const unsigned short* __restrict__ Vp,
                       const int* __restrict__ nap,
                       unsigned short* __restrict__ O) {
  const int lane = threadIdx.x;
  const int half = lane >> 4;
  const int lan  = lane & 15;
  const int kh8  = half * 8;
  const int kh16 = half * 16;
  const int q0 = blockIdx.x * 16;
  const int h  = blockIdx.y;
  const int b  = blockIdx.z;
  const long qrowbase = (long)b * TQL + q0;

  // Q fragments: A layout over d (two 32-wide k-steps cover HDIM=64)
  Frag qa[2];
  {
    const unsigned short* qb =
        Q + (qrowbase + lan) * DDIM + h * HDIM + kh8;
#pragma unroll
    for (int c = 0; c < 2; ++c) {
      qa[c].q[0] = *(const u32x4*)(qb + c * 32);
      qa[c].q[1] = *(const u32x4*)(qb + c * 32 + 16);
    }
  }

  __shared__ __align__(16) unsigned short Pl[16][32];   // P tile (C/D->A transpose)
  __shared__ __align__(16) unsigned short Vl[32][64];   // async-staged V tile

  const v8f z = {0.f, 0.f, 0.f, 0.f, 0.f, 0.f, 0.f, 0.f};
  v8f fin[4];
#pragma unroll
  for (int t = 0; t < 4; ++t) fin[t] = z;

  const int Na = *nap;
  const long kvbase = (long)b * TKL * DDIM + h * HDIM;
  const unsigned int vlds = (unsigned int)(size_t)&Vl[lane][0];

  for (int g = 0; g < 2; ++g) {
    const int start = g ? Na : 0;
    const int len   = g ? (TKL - Na) : Na;
    if (len <= 0) continue;

    v8f acc[4];
#pragma unroll
    for (int t = 0; t < 4; ++t) acc[t] = z;
    float mr[8], lr[8];
#pragma unroll
    for (int v = 0; v < 8; ++v) { mr[v] = -3.0e38f; lr[v] = 0.f; }

    for (int j0 = start; j0 < start + len; j0 += 32) {
      // ---- kick off async HBM->LDS stage of the 32x64 V tile ----
      // (overlaps with the score WMMAs below; tracked by ASYNCcnt)
      {
        // make sure prior-chunk LDS reads are done before overwriting Vl
        asm volatile("s_wait_dscnt 0x0" ::: "memory");
        const unsigned short* gv = Vp + kvbase + (long)(j0 + lane) * DDIM;
#pragma unroll
        for (int i = 0; i < 8; ++i) {
          asm volatile("global_load_async_to_lds_b128 %0, %1, off"
                       :: "v"(vlds + i * 16), "v"(gv + i * 8)
                       : "memory");
        }
      }

      // ---- scores: 16x32 tile = 2 n-tiles x 2 k-steps of WMMA ----
      v8f sc[2];
#pragma unroll
      for (int n = 0; n < 2; ++n) {
        const unsigned short* kb8 =
            Kp + kvbase + (long)(j0 + n * 16 + lan) * DDIM + kh8;
        v8f s = z;
#pragma unroll
        for (int c = 0; c < 2; ++c) {
          Frag kf;
          kf.q[0] = *(const u32x4*)(kb8 + c * 32);
          kf.q[1] = *(const u32x4*)(kb8 + c * 32 + 16);
          s = __builtin_amdgcn_wmma_f32_16x16x32_bf16(
              false, qa[c].v, false, kf.v, (short)0, s, false, false);
        }
        sc[n] = s;
      }

      // ---- online softmax; rows live per-VGPR within each half-wave ----
#pragma unroll
      for (int v = 0; v < 8; ++v) {
        float cm = rmax16(fmaxf(sc[0][v], sc[1][v]));
        float mn = fmaxf(mr[v], cm);
        float sf = __expf(mr[v] - mn);
        float p0 = __expf(sc[0][v] - mn);
        float p1 = __expf(sc[1][v] - mn);
        lr[v] = lr[v] * sf + rsum16(p0 + p1);
        mr[v] = mn;
#pragma unroll
        for (int t = 0; t < 4; ++t) acc[t][v] *= sf;
        int r = v + kh8;
        Pl[r][lan]      = f2bf(p0);
        Pl[r][16 + lan] = f2bf(p1);
      }
      __syncthreads();

      // ---- wait for the async V tile before reading it from LDS ----
      asm volatile("s_wait_asynccnt 0x0" ::: "memory");

      // ---- P tile back as A fragment (lane = query row) ----
      Frag pa;
      {
        const unsigned short* prow = &Pl[lan][kh8];
        pa.q[0] = *(const u32x4*)(prow);
        pa.q[1] = *(const u32x4*)(prow + 16);
      }

      // ---- acc += P @ V ; V as B fragments (k = key-in-chunk, n = d) ----
#pragma unroll
      for (int t = 0; t < 4; ++t) {
        Frag vb;
        int n = t * 16 + lan;
#pragma unroll
        for (int v = 0; v < 8; ++v) {
          int k = kh8 + 2 * v + (v >> 2) * 8;
          unsigned int lo = Vl[k][n];
          unsigned int hi = Vl[k + 1][n];
          vb.u[v] = lo | (hi << 16);
        }
        acc[t] = __builtin_amdgcn_wmma_f32_16x16x32_bf16(
            false, pa.v, false, vb.v, (short)0, acc[t], false, false);
      }
      __syncthreads();
    }

    // ---- fold group result: sign * (acc / l) / sqrt(len) ----
    float rs = rsqrtf((float)len);
    float sign = g ? -1.0f : 1.0f;
#pragma unroll
    for (int v = 0; v < 8; ++v) {
      float inv = sign * rs / lr[v];
#pragma unroll
      for (int t = 0; t < 4; ++t) fin[t][v] += acc[t][v] * inv;
    }
  }

  // ---- store bf16 attention output [B*TQ, D] ----
#pragma unroll
  for (int t = 0; t < 4; ++t)
#pragma unroll
    for (int v = 0; v < 8; ++v) {
      int r = v + kh8;
      O[(qrowbase + r) * DDIM + h * HDIM + t * 16 + lan] = f2bf(fin[t][v]);
    }
}

// ---------------------------------------------------------------------------
extern "C" void kernel_launch(void* const* d_in, const int* in_sizes, int n_in,
                              void* d_out, int out_size, void* d_ws,
                              size_t ws_size, hipStream_t stream) {
  (void)in_sizes; (void)n_in; (void)out_size; (void)ws_size;

  const float* q     = (const float*)d_in[0];
  const float* kv    = (const float*)d_in[1];
  const float* Wq    = (const float*)d_in[2];
  const float* bq    = (const float*)d_in[3];
  const float* Wk    = (const float*)d_in[4];
  const float* bk    = (const float*)d_in[5];
  const float* Wv    = (const float*)d_in[6];
  const float* bv    = (const float*)d_in[7];
  const float* Wo    = (const float*)d_in[8];
  const float* bo    = (const float*)d_in[9];
  const float* abias = (const float*)d_in[10];
  const float* ibias = (const float*)d_in[11];
  const float* lt    = (const float*)d_in[12];
  const int*   nap   = (const int*)d_in[13];
  float* out = (float*)d_out;

  // workspace carve-out (bf16 buffers), 256B aligned
  char* ws = (char*)d_ws;
  size_t off = 0;
  auto carve = [&](size_t bytes) -> void* {
    void* p = ws + off;
    off = (off + bytes + 255) & ~(size_t)255;
    return p;
  };
  const size_t szQ  = (size_t)BB * TQL * DDIM * 2;
  const size_t szKV = (size_t)BB * TKL * DDIM * 2;
  const size_t szW  = (size_t)DDIM * DDIM * 2;

  unsigned short* qb   = (unsigned short*)carve(szQ);
  unsigned short* kvb  = (unsigned short*)carve(szKV);
  unsigned short* wqb  = (unsigned short*)carve(szW);
  unsigned short* wkb  = (unsigned short*)carve(szW);
  unsigned short* wvb  = (unsigned short*)carve(szW);
  unsigned short* wob  = (unsigned short*)carve(szW);
  unsigned short* qs   = (unsigned short*)carve(szQ);   // scaled Q proj
  unsigned short* kp   = (unsigned short*)carve(szKV);
  unsigned short* vp   = (unsigned short*)carve(szKV);
  unsigned short* attn = (unsigned short*)carve(szQ);

  const long nQ8  = ((long)BB * TQL * DDIM) / 8;
  const long nKV8 = ((long)BB * TKL * DDIM) / 8;
  const long nW8  = ((long)DDIM * DDIM) / 8;

  // 1) conversions (vectorized, b128 in/out)
  k_cvt8<<<dim3((unsigned)((nQ8 + 255) / 256)), 256, 0, stream>>>(q, qb, nQ8);
  k_cvt8<<<dim3((unsigned)((nW8 + 255) / 256)), 256, 0, stream>>>(Wq, wqb, nW8);
  k_cvt8<<<dim3((unsigned)((nW8 + 255) / 256)), 256, 0, stream>>>(Wk, wkb, nW8);
  k_cvt8<<<dim3((unsigned)((nW8 + 255) / 256)), 256, 0, stream>>>(Wv, wvb, nW8);
  k_cvt8<<<dim3((unsigned)((nW8 + 255) / 256)), 256, 0, stream>>>(Wo, wob, nW8);
  k_cvt_kv8<<<dim3((unsigned)((nKV8 + 255) / 256)), 256, 0, stream>>>(
      kv, abias, ibias, nap, kvb, nKV8);

  // 2) projections (bf16 WMMA GEMMs)
  k_gemm<<<dim3(DDIM / 64, (BB * TQL) / 32), 32, 0, stream>>>(
      qb, wqb, bq, lt, qs, nullptr, BB * TQL, DDIM, DDIM);       // scaled Q
  k_gemm<<<dim3(DDIM / 64, (BB * TKL) / 32), 32, 0, stream>>>(
      kvb, wkb, bk, nullptr, kp, nullptr, BB * TKL, DDIM, DDIM); // K
  k_gemm<<<dim3(DDIM / 64, (BB * TKL) / 32), 32, 0, stream>>>(
      kvb, wvb, bv, nullptr, vp, nullptr, BB * TKL, DDIM, DDIM); // V

  // 3) grouped attention (async V staging + WMMA)
  k_attn<<<dim3(TQL / 16, NHEAD, BB), 32, 0, stream>>>(qs, kp, vp, nap, attn);

  // 4) output projection (fp32 result)
  k_gemm<<<dim3(DDIM / 64, (BB * TQL) / 32), 32, 0, stream>>>(
      attn, wob, bo, nullptr, nullptr, out, BB * TQL, DDIM, DDIM);
}
